// MambaLayer_33852932227255
// MI455X (gfx1250) — matrix operand order repaired and checked by reading, not compiled, not run on hardware
//
#include <hip/hip_runtime.h>

// ---------------- problem constants (match reference) ----------------
#define B_      2
#define L_      1024
#define DM      1024
#define D_INNER 2048
#define D_STATE 64
#define D_CONV  4
#define DT_RANK 64
#define NTOK    (B_ * L_)              // 2048 tokens

// ---------------- CDNA5 WMMA types ----------------
typedef __attribute__((ext_vector_type(16))) __bf16 v16bf;
typedef __attribute__((ext_vector_type(8)))  float  v8f;

// float -> bf16 round-to-nearest-even, as raw u16 (pure bit ops)
static __device__ __forceinline__ unsigned short f2bf_u16(float f) {
    union { float f; unsigned u; } v; v.f = f;
    unsigned r = v.u + 0x7FFFu + ((v.u >> 16) & 1u);
    return (unsigned short)(r >> 16);
}
static __device__ __forceinline__ unsigned pack2(float a, float b) {
    return (unsigned)f2bf_u16(a) | ((unsigned)f2bf_u16(b) << 16);
}
static __device__ __forceinline__ float silu(float v) {
    return v / (1.f + __expf(-v));
}

// ---------------- 0) bulk f32 -> bf16 convert (weights; 4 elems/thread) ----------------
__global__ __launch_bounds__(256) void f32_to_bf16_kernel(const float* __restrict__ src,
                                                          __bf16* __restrict__ dst, int n4) {
    int i = blockIdx.x * 256 + threadIdx.x;
    if (i >= n4) return;
    float4 v = ((const float4*)src)[i];
    uint2 p; p.x = pack2(v.x, v.y); p.y = pack2(v.z, v.w);
    ((uint2*)dst)[i] = p;
}

// ---------------- 1) LayerNorm: x[NTOK,DM] -> xn bf16 ----------------
__global__ __launch_bounds__(256) void ln_kernel(const float* __restrict__ x,
                                                 const float* __restrict__ w,
                                                 const float* __restrict__ b,
                                                 __bf16* __restrict__ out) {
    const int row = blockIdx.x;
    const float* xr = x + (size_t)row * DM;
    float s = 0.f, ss = 0.f;
    for (int i = threadIdx.x; i < DM; i += 256) { float v = xr[i]; s += v; ss += v * v; }
#pragma unroll
    for (int off = 16; off > 0; off >>= 1) { s += __shfl_xor(s, off); ss += __shfl_xor(ss, off); }
    __shared__ float red[16];
    __shared__ float stats[2];
    const int wave = threadIdx.x >> 5, lane = threadIdx.x & 31;
    if (lane == 0) { red[wave] = s; red[8 + wave] = ss; }
    __syncthreads();
    if (threadIdx.x == 0) {
        float ts = 0.f, tss = 0.f;
        for (int i = 0; i < 8; ++i) { ts += red[i]; tss += red[8 + i]; }
        float mu  = ts / (float)DM;
        float var = tss / (float)DM - mu * mu;
        stats[0] = mu; stats[1] = rsqrtf(var + 1e-5f);
    }
    __syncthreads();
    const float mu = stats[0], rs = stats[1];
    // vectorized epilogue: 4 contiguous elems per thread (DM = 1024 = 256*4)
    const int i = threadIdx.x << 2;
    float4 xv = *(const float4*)(xr + i);
    float4 wv = *(const float4*)(w + i);
    float4 bv = *(const float4*)(b + i);
    uint2 p;
    p.x = pack2((xv.x - mu) * rs * wv.x + bv.x, (xv.y - mu) * rs * wv.y + bv.y);
    p.y = pack2((xv.z - mu) * rs * wv.z + bv.z, (xv.w - mu) * rs * wv.w + bv.w);
    *(uint2*)(out + (size_t)row * DM + i) = p;
}

// ---------------- 2) Tiled pure-bf16 WMMA GEMM ----------------
// C[M,N](f32) = A[M,K](bf16, lda) * B[K,N](bf16, ldb).
// Block: 256 threads = 8 waves (4 M-strips x 2 N-strips); block tile 128x64;
// wave tile 32x32 -> 4 accumulators, 4 WMMAs per K-step.
// REQUIRES: M%128==0, N%64==0, K%32==0, lda%8==0, ldb%8==0 (16B-aligned rows).
#define TM  128
#define TN  64
#define TK  32
#define TKP 40   // LDS row pitch in bf16 (80 B = 5*16: aligned b128 chunks + bank skew)

__global__ __launch_bounds__(256) void gemm_bf16_wmma(const __bf16* __restrict__ A,
                                                      const __bf16* __restrict__ B,
                                                      float* __restrict__ C,
                                                      int M, int N, int K,
                                                      int lda, int ldb, int ldc) {
    __shared__ __bf16 As[TM][TKP];   // A tile, row-major (M x K)
    __shared__ __bf16 Bs[TN][TKP];   // B tile, transposed (N x K): contiguous-K lane loads

    const int tid  = threadIdx.x;
    const int wave = tid >> 5;
    const int lane = tid & 31;
    const int wm   = wave >> 1;       // 0..3 : 32-row strip
    const int wn   = wave & 1;        // 0..1 : 32-col strip
    const int row  = lane & 15;
    const int hi   = lane >> 4;

    const size_t block_m = (size_t)blockIdx.y * TM;
    const size_t block_n = (size_t)blockIdx.x * TN;

    v8f acc00 = {}, acc01 = {}, acc10 = {}, acc11 = {};

    for (int k0 = 0; k0 < K; k0 += TK) {
        // ---- stage A tile (128x32 bf16): 512 x 8-elem groups, 2 per thread, pure b128 copy ----
#pragma unroll
        for (int it = 0; it < 2; ++it) {
            const int i = tid + it * 256;     // static trip count: no exec guard
            const int m = i >> 2;             // 4 groups per 32-wide row
            const int k = (i & 3) << 3;
            *(uint4*)&As[m][k] = *(const uint4*)(A + (block_m + m) * lda + (k0 + k));
        }
        // ---- stage B tile transposed (Bs[n][k] = B[k0+k][block_n+n]): 256 groups, 1/thread ----
        {
            const int k = tid >> 3;           // 8 groups per 64-wide B row
            const int n = (tid & 7) << 3;
            union { uint4 q; __bf16 e[8]; } wv;
            wv.q = *(const uint4*)(B + (size_t)(k0 + k) * ldb + (block_n + n));
#pragma unroll
            for (int j = 0; j < 8; ++j) Bs[n + j][k] = wv.e[j];
        }
        __syncthreads();

        // ---- A fragments (16x32): half 0 -> K {0..7,16..23}; half 1 -> K {8..15,24..31} ----
        v16bf a0, a1;
        const int am0 = wm * 32 + row;
        const int am1 = am0 + 16;
#pragma unroll
        for (int t = 0; t < 8; ++t) {
            a0[t]     = As[am0][hi * 8 + t];
            a0[8 + t] = As[am0][16 + hi * 8 + t];
            a1[t]     = As[am1][hi * 8 + t];
            a1[8 + t] = As[am1][16 + hi * 8 + t];
        }
        // ---- B fragments (32x16): half 0 -> K {0..7,16..23}; half 1 -> K {8..15,24..31} ----
        v16bf b0, b1;
        const int bn0 = wn * 32 + row;
        const int bn1 = bn0 + 16;
#pragma unroll
        for (int t = 0; t < 8; ++t) {
            b0[t]     = Bs[bn0][hi * 8 + t];
            b0[8 + t] = Bs[bn0][16 + hi * 8 + t];
            b1[t]     = Bs[bn1][hi * 8 + t];
            b1[8 + t] = Bs[bn1][16 + hi * 8 + t];
        }
        acc00 = __builtin_amdgcn_wmma_f32_16x16x32_bf16(false, a0, false, b0, (short)0, acc00, false, false);
        acc01 = __builtin_amdgcn_wmma_f32_16x16x32_bf16(false, a0, false, b1, (short)0, acc01, false, false);
        acc10 = __builtin_amdgcn_wmma_f32_16x16x32_bf16(false, a1, false, b0, (short)0, acc10, false, false);
        acc11 = __builtin_amdgcn_wmma_f32_16x16x32_bf16(false, a1, false, b1, (short)0, acc11, false, false);
        __syncthreads();
    }

    // ---- store: lanes 0-15 hold M=r, lanes 16-31 hold M=8+r; N = lane&15 ----
    const size_t cm0 = block_m + wm * 32 + hi * 8;
    const size_t cm1 = cm0 + 16;
    const size_t cn0 = block_n + wn * 32 + row;
    const size_t cn1 = cn0 + 16;
#pragma unroll
    for (int r = 0; r < 8; ++r) {
        C[(cm0 + r) * ldc + cn0] = acc00[r];
        C[(cm0 + r) * ldc + cn1] = acc01[r];
        C[(cm1 + r) * ldc + cn0] = acc10[r];
        C[(cm1 + r) * ldc + cn1] = acc11[r];
    }
}

// ---------------- 3) depthwise causal conv1d + bias + SiLU (4 channels/thread) ----------------
// u_f32 for scan/gate, u_bf16 for the next GEMM's A operand.
__global__ __launch_bounds__(256) void conv_silu_kernel(const float* __restrict__ xz,
                                                        const float* __restrict__ cw,
                                                        const float* __restrict__ cb,
                                                        float* __restrict__ u,
                                                        __bf16* __restrict__ u_bf) {
    const size_t g = (size_t)blockIdx.x * 256 + threadIdx.x;   // over NTOK*D_INNER/4
    if (g >= (size_t)NTOK * D_INNER / 4) return;
    const int c = (int)((g << 2) % D_INNER);
    const int t = (int)((g << 2) / D_INNER);   // b*L + l
    const int l = t % L_;
    float4 acc = *(const float4*)(cb + c);
#pragma unroll
    for (int j = 0; j < D_CONV; ++j) {
        const int ls = l - (D_CONV - 1) + j;
        if (ls >= 0) {
            const float4 xv = *(const float4*)(xz + (size_t)(t - (D_CONV - 1) + j) * (2 * D_INNER) + c);
            acc.x += xv.x * cw[(c + 0) * D_CONV + j];
            acc.y += xv.y * cw[(c + 1) * D_CONV + j];
            acc.z += xv.z * cw[(c + 2) * D_CONV + j];
            acc.w += xv.w * cw[(c + 3) * D_CONV + j];
        }
    }
    float4 r; r.x = silu(acc.x); r.y = silu(acc.y); r.z = silu(acc.z); r.w = silu(acc.w);
    ((float4*)u)[g] = r;
    uint2 p; p.x = pack2(r.x, r.y); p.y = pack2(r.z, r.w);
    ((uint2*)u_bf)[g] = p;
}

// ---------------- 4) extract dt_in = proj[:, :64] as bf16 [NTOK, 64] ----------------
__global__ __launch_bounds__(256) void extract_dtin_kernel(const float* __restrict__ proj,
                                                           __bf16* __restrict__ dtin) {
    const int i = blockIdx.x * 256 + threadIdx.x;   // over NTOK*16 groups of 4
    if (i >= NTOK * 16) return;
    const int r = i >> 4, g = (i & 15) << 2;
    const float4 v = *(const float4*)(proj + (size_t)r * (DT_RANK + 2 * D_STATE) + g);
    uint2 p; p.x = pack2(v.x, v.y); p.y = pack2(v.z, v.w);
    *(uint2*)(dtin + (size_t)r * DT_RANK + g) = p;
}

// ---------------- 5) dt bias + softplus, 4-wide in place ----------------
__global__ __launch_bounds__(256) void bias_softplus_kernel(float* __restrict__ dt,
                                                            const float* __restrict__ bdt) {
    const size_t g = (size_t)blockIdx.x * 256 + threadIdx.x;
    if (g >= (size_t)NTOK * D_INNER / 4) return;
    const int c = (int)((g << 2) % D_INNER);
    float4 v = ((const float4*)dt)[g];
    const float4 b = *(const float4*)(bdt + c);
    v.x += b.x; v.y += b.y; v.z += b.z; v.w += b.w;
    v.x = (v.x > 20.f) ? v.x : log1pf(__expf(v.x));
    v.y = (v.y > 20.f) ? v.y : log1pf(__expf(v.y));
    v.z = (v.z > 20.f) ? v.z : log1pf(__expf(v.z));
    v.w = (v.w > 20.f) ? v.w : log1pf(__expf(v.w));
    ((float4*)dt)[g] = v;
}

// ---------------- 6) selective scan: one wave32 per (b, d) channel ----------------
__global__ __launch_bounds__(256) void scan_kernel(const float* __restrict__ dt,
                                                   const float* __restrict__ u,
                                                   const float* __restrict__ proj,
                                                   const float* __restrict__ A_log,
                                                   float* __restrict__ ys) {
    const int gwave = (int)((blockIdx.x * 256 + threadIdx.x) >> 5);  // 0 .. B_*D_INNER-1
    const int lane  = threadIdx.x & 31;
    const int b = gwave / D_INNER;
    const int d = gwave % D_INNER;
    const int s0 = lane * 2, s1 = s0 + 1;
    const int ldp = DT_RANK + 2 * D_STATE;   // 192

    const float A0 = -__expf(A_log[(size_t)d * D_STATE + s0]);
    const float A1 = -__expf(A_log[(size_t)d * D_STATE + s1]);
    float h0 = 0.f, h1 = 0.f;

    for (int l = 0; l < L_; ++l) {
        const size_t row = (size_t)(b * L_ + l);
        const float dtv = dt[row * D_INNER + d];
        const float uv  = u[row * D_INNER + d];
        const float* pr = proj + row * ldp;
        const float Bm0 = pr[DT_RANK + s0],           Bm1 = pr[DT_RANK + s1];
        const float Cm0 = pr[DT_RANK + D_STATE + s0], Cm1 = pr[DT_RANK + D_STATE + s1];

        // speculative prefetch of next timestep (architecturally safe)
        __builtin_prefetch(dt + (row + 1) * D_INNER + d, 0, 3);
        __builtin_prefetch(proj + (row + 1) * ldp + DT_RANK + s0, 0, 3);

        const float du = dtv * uv;
        h0 = __expf(dtv * A0) * h0 + du * Bm0;
        h1 = __expf(dtv * A1) * h1 + du * Bm1;
        float y = h0 * Cm0 + h1 * Cm1;
#pragma unroll
        for (int off = 16; off > 0; off >>= 1) y += __shfl_xor(y, off);
        if (lane == 0) ys[row * D_INNER + d] = y;
    }
}

// ---------------- 7) gate: y_bf = bf16((ys + u*D) * silu(z)), 4-wide ----------------
__global__ __launch_bounds__(256) void gate_kernel(const float* __restrict__ ys,
                                                   const float* __restrict__ u,
                                                   const float* __restrict__ Dv,
                                                   const float* __restrict__ xz,
                                                   __bf16* __restrict__ y_bf) {
    const size_t g = (size_t)blockIdx.x * 256 + threadIdx.x;
    if (g >= (size_t)NTOK * D_INNER / 4) return;
    const int c = (int)((g << 2) % D_INNER);
    const int t = (int)((g << 2) / D_INNER);
    const float4 zv = *(const float4*)(xz + (size_t)t * (2 * D_INNER) + D_INNER + c);
    const float4 yv = ((const float4*)ys)[g];
    const float4 uv = ((const float4*)u)[g];
    const float4 dv = *(const float4*)(Dv + c);
    float4 r;
    r.x = (yv.x + uv.x * dv.x) * silu(zv.x);
    r.y = (yv.y + uv.y * dv.y) * silu(zv.y);
    r.z = (yv.z + uv.z * dv.z) * silu(zv.z);
    r.w = (yv.w + uv.w * dv.w) * silu(zv.w);
    uint2 p; p.x = pack2(r.x, r.y); p.y = pack2(r.z, r.w);
    ((uint2*)y_bf)[g] = p;
}

// ---------------- host orchestration ----------------
extern "C" void kernel_launch(void* const* d_in, const int* in_sizes, int n_in,
                              void* d_out, int out_size, void* d_ws, size_t ws_size,
                              hipStream_t stream) {
    const float* x      = (const float*)d_in[0];
    const float* norm_w = (const float*)d_in[1];
    const float* norm_b = (const float*)d_in[2];
    const float* W_in   = (const float*)d_in[3];
    const float* conv_w = (const float*)d_in[4];
    const float* conv_b = (const float*)d_in[5];
    const float* W_xprj = (const float*)d_in[6];
    const float* W_dt   = (const float*)d_in[7];
    const float* b_dt   = (const float*)d_in[8];
    const float* A_log  = (const float*)d_in[9];
    const float* Dvec   = (const float*)d_in[10];
    const float* W_out  = (const float*)d_in[11];
    float* out = (float*)d_out;

    const int ldp = DT_RANK + 2 * D_STATE;  // 192

    // ---- workspace carve: f32 region first, then 16B-aligned bf16 region ----
    float* ws_f = (float*)d_ws;
    float* xz   = ws_f;                                  // [NTOK, 4096]
    float* proj = xz   + (size_t)NTOK * 2 * D_INNER;     // [NTOK, 192]
    float* dt   = proj + (size_t)NTOK * ldp;             // [NTOK, 2048]
    float* u    = dt   + (size_t)NTOK * D_INNER;         // [NTOK, 2048]
    float* ys   = u    + (size_t)NTOK * D_INNER;         // [NTOK, 2048]
    __bf16* xn_bf   = (__bf16*)(ys + (size_t)NTOK * D_INNER);
    __bf16* u_bf    = xn_bf   + (size_t)NTOK * DM;
    __bf16* y_bf    = u_bf    + (size_t)NTOK * D_INNER;
    __bf16* dtin_bf = y_bf    + (size_t)NTOK * D_INNER;
    __bf16* Win_bf  = dtin_bf + (size_t)NTOK * DT_RANK;
    __bf16* Wxp_bf  = Win_bf  + (size_t)DM * 2 * D_INNER;
    __bf16* Wdt_bf  = Wxp_bf  + (size_t)D_INNER * ldp;
    __bf16* Wout_bf = Wdt_bf  + (size_t)DT_RANK * D_INNER;

    const int nel4 = NTOK * D_INNER / 4;   // 4-wide elementwise grids

    // 0) one-shot weight conversion to bf16
    f32_to_bf16_kernel<<<(DM * 2 * D_INNER / 4 + 255) / 256, 256, 0, stream>>>(W_in,  Win_bf,  DM * 2 * D_INNER / 4);
    f32_to_bf16_kernel<<<(D_INNER * ldp / 4 + 255) / 256, 256, 0, stream>>>(W_xprj, Wxp_bf,  D_INNER * ldp / 4);
    f32_to_bf16_kernel<<<(DT_RANK * D_INNER / 4 + 255) / 256, 256, 0, stream>>>(W_dt, Wdt_bf, DT_RANK * D_INNER / 4);
    f32_to_bf16_kernel<<<(D_INNER * DM / 4 + 255) / 256, 256, 0, stream>>>(W_out, Wout_bf,  D_INNER * DM / 4);

    // 1) LayerNorm -> bf16
    ln_kernel<<<NTOK, 256, 0, stream>>>(x, norm_w, norm_b, xn_bf);

    // 2) xz = xn @ W_in    [2048 x 4096, K=1024]
    gemm_bf16_wmma<<<dim3((2 * D_INNER) / TN, NTOK / TM), 256, 0, stream>>>(
        xn_bf, Win_bf, xz, NTOK, 2 * D_INNER, DM, DM, 2 * D_INNER, 2 * D_INNER);

    // 3) depthwise causal conv + SiLU -> u (f32) + u_bf (bf16)
    conv_silu_kernel<<<(nel4 + 255) / 256, 256, 0, stream>>>(xz, conv_w, conv_b, u, u_bf);

    // 4) proj = u @ W_xproj  [2048 x 192, K=2048]
    gemm_bf16_wmma<<<dim3(ldp / TN, NTOK / TM), 256, 0, stream>>>(
        u_bf, Wxp_bf, proj, NTOK, ldp, D_INNER, D_INNER, ldp, ldp);

    // 5) dt_in slice -> bf16 [NTOK, 64]
    extract_dtin_kernel<<<(NTOK * 16 + 255) / 256, 256, 0, stream>>>(proj, dtin_bf);

    // 6) dt_lin = dt_in @ W_dt  [2048 x 2048, K=64]
    gemm_bf16_wmma<<<dim3(D_INNER / TN, NTOK / TM), 256, 0, stream>>>(
        dtin_bf, Wdt_bf, dt, NTOK, D_INNER, DT_RANK, DT_RANK, D_INNER, D_INNER);

    // 7) dt = softplus(dt_lin + b_dt)
    bias_softplus_kernel<<<(nel4 + 255) / 256, 256, 0, stream>>>(dt, b_dt);

    // 8) selective scan: 4096 waves (one per channel), 8 waves / block
    scan_kernel<<<(B_ * D_INNER) / 8, 256, 0, stream>>>(dt, u, proj, A_log, ys);

    // 9) gate -> y_bf
    gate_kernel<<<(nel4 + 255) / 256, 256, 0, stream>>>(ys, u, Dvec, xz, y_bf);

    // 10) out = y @ W_out  [2048 x 1024, K=2048]
    gemm_bf16_wmma<<<dim3(DM / TN, NTOK / TM), 256, 0, stream>>>(
        y_bf, Wout_bf, out, NTOK, DM, D_INNER, D_INNER, DM, DM);
}
